// Encoder_54288386621878
// MI455X (gfx1250) — matrix operand order repaired and checked by reading, not compiled, not run on hardware
//
#include <hip/hip_runtime.h>
#include <math.h>

typedef __attribute__((ext_vector_type(16))) _Float16 v16h;
typedef __attribute__((ext_vector_type(8)))  _Float16 v8h;
typedef __attribute__((ext_vector_type(8)))  float    v8f;

#define B_   16
#define T_   16384
#define I_   64
#define H_   128
#define KTOT (I_ + H_)     // 192 combined K (x | h)
#define WAVES 16
#define THREADS (WAVES * 32)

// Split workgroup barrier with DS-only dependency wait: unlike __syncthreads()
// (which emits s_wait_loadcnt_dscnt 0x0 and forces our x-prefetch global load
// to complete at every barrier), this waits only DScnt, keeping the prefetch
// outstanding across the barrier.
#define WG_BARRIER_DS()                                           \
    asm volatile("s_wait_dscnt 0x0\n\t"                           \
                 "s_barrier_signal -1\n\t"                        \
                 "s_barrier_wait -1" ::: "memory")

__device__ __forceinline__ float fast_rcp(float x) {
    return __builtin_amdgcn_rcpf(x);
}
__device__ __forceinline__ float sigmoid_f(float x) {
    return fast_rcp(1.0f + __expf(-x));            // v_exp_f32 + v_rcp_f32
}
__device__ __forceinline__ float tanh_f(float x) {
    const float e2 = __expf(2.0f * x);
    return (e2 - 1.0f) * fast_rcp(e2 + 1.0f);
}

// ---------------------------------------------------------------------------
// Phase 1: persistent single-workgroup LSTM scan.
// 16 waves; wave w owns N-tiles {w, w+16} of the 32 (16-wide) gate tiles:
//   waves 0..7  -> i-gate (tile w) and g-gate (tile w+16), cols [16w,16w+16)
//   waves 8..15 -> f-gate (tile w) and o-gate (tile w+16), cols [16(w-8),...)
// waves 0..7 publish p = sigmoid(i)*tanh(g) pre-barrier; waves 8..15 keep
// sigmoid(f)/sigmoid(o) in registers and own the c/h state update, so the
// post-barrier critical path is just one FMA + one tanh.
// Weight B-fragments live in VGPRs for the whole kernel (12 x v16h per wave).
// ---------------------------------------------------------------------------
__global__ __launch_bounds__(THREADS, 1)
void lstm_scan_kernel(const float* __restrict__ x,
                      const float* __restrict__ Wih,
                      const float* __restrict__ Whh,
                      const float* __restrict__ bih,
                      const float* __restrict__ bhh,
                      float* __restrict__ out,    // y raw (pre-norm) [B,T,H]
                      float* __restrict__ hn,     // [B,H]
                      float* __restrict__ cn,     // [B,H]
                      float* __restrict__ gsum,   // [H]
                      float* __restrict__ gsumsq) // [H]
{
    __shared__ _Float16 Abuf[B_ * KTOT];       // [16 x (64 x | 128 h)] f16, 6 KB
    __shared__ float    exch[8][B_][16];       // i*g products, 8 KB

    const int tid  = threadIdx.x;
    const int w    = tid >> 5;     // wave id 0..15
    const int lane = tid & 31;
    const int mr   = lane & 15;    // column-within-tile / row selector
    const int hi   = lane >> 4;    // lane half

    // ---- build B fragments in registers (B[k,n] = W[n,k]) -----------------
    // B 32x16 f16 layout: lane holds col n = l&15, K = kc*32 + hi*8 + (j&7) + (j>>3)*16.
    v16h Bf[2][6];
#pragma unroll
    for (int ti = 0; ti < 2; ++ti) {
        const int nt = (ti == 0) ? w : (w + 16);
        const int r  = nt * 16 + mr;           // row of combined W [512 x 192]
#pragma unroll
        for (int kc = 0; kc < 6; ++kc) {
            const int k0 = kc * 32 + hi * 8;
#pragma unroll
            for (int j = 0; j < 16; ++j) {
                const int k = k0 + (j & 7) + ((j >> 3) << 4);
                const float v = (k < I_) ? Wih[r * I_ + k]
                                         : Whh[r * H_ + (k - I_)];
                Bf[ti][kc][j] = (_Float16)v;
            }
        }
    }
    const float bias0 = bih[w * 16 + mr]        + bhh[w * 16 + mr];
    const float bias1 = bih[(w + 16) * 16 + mr] + bhh[(w + 16) * 16 + mr];

    // ---- zero h region of A buffer (h_{-1} = 0) ---------------------------
    for (int idx = tid; idx < B_ * H_; idx += THREADS) {
        const int rrow = idx >> 7;
        const int hcol = idx & (H_ - 1);
        Abuf[rrow * KTOT + I_ + hcol] = (_Float16)0.0f;
    }

    // ---- x prefetch: each thread owns 2 floats of x_t ---------------------
    const int xb = tid >> 5;             // batch row 0..15
    const int xi = (tid & 31) * 2;       // input col 0..62
    float2 px = *(const float2*)(x + ((size_t)xb * T_ + 0) * I_ + xi);

    v8f cstate = {0.f,0.f,0.f,0.f,0.f,0.f,0.f,0.f};
    v8f hstate = {0.f,0.f,0.f,0.f,0.f,0.f,0.f,0.f};
    v8f ssum   = {0.f,0.f,0.f,0.f,0.f,0.f,0.f,0.f};
    v8f ssq    = {0.f,0.f,0.f,0.f,0.f,0.f,0.f,0.f};

    for (int t = 0; t < T_; ++t) {
        // stage prefetched x_t into A buffer (f32 -> f16)
        Abuf[xb * KTOT + xi]     = (_Float16)px.x;
        Abuf[xb * KTOT + xi + 1] = (_Float16)px.y;
        // prefetch x_{t+1}: issued before the barrier, consumed at the next
        // loop top -> a full step of latency coverage (the DS-only barrier
        // does NOT force it to complete).
        const int tn = (t + 1 < T_) ? (t + 1) : t;
        px = *(const float2*)(x + ((size_t)xb * T_ + tn) * I_ + xi);

        WG_BARRIER_DS();   // B1: x_t and h_{t-1} visible in Abuf

        // A-fragment LDS loads (scheduler double-buffers across K-chunks).
        v8h alo[6], ahi[6];
#pragma unroll
        for (int kc = 0; kc < 6; ++kc) {
            // A 16x32 f16 layout: lane half hi -> K runs at hi*8 and 16+hi*8
            const _Float16* ap = &Abuf[mr * KTOT + kc * 32 + hi * 8];
            alo[kc] = *(const v8h*)(ap);
            ahi[kc] = *(const v8h*)(ap + 16);
        }

        v8f acc0, acc1;
#pragma unroll
        for (int j = 0; j < 8; ++j) { acc0[j] = bias0; acc1[j] = bias1; }

#pragma unroll
        for (int kc = 0; kc < 6; ++kc) {
            v16h a;
#pragma unroll
            for (int j = 0; j < 8; ++j) { a[j] = alo[kc][j]; a[8 + j] = ahi[kc][j]; }
            acc0 = __builtin_amdgcn_wmma_f32_16x16x32_f16(
                       false, a, false, Bf[0][kc], (short)0, acc0, false, false);
            acc1 = __builtin_amdgcn_wmma_f32_16x16x32_f16(
                       false, a, false, Bf[1][kc], (short)0, acc1, false, false);
        }

        float sf[8], so[8];
        if (w < 8) {    // i/g gates: publish p = sigmoid(i)*tanh(g) to LDS
#pragma unroll
            for (int v = 0; v < 8; ++v) {
                const int m = v + 8 * hi;
                exch[w][m][mr] = sigmoid_f(acc0[v]) * tanh_f(acc1[v]);
            }
        } else {        // f/o gates: keep sigmoid(f), sigmoid(o) in registers
#pragma unroll
            for (int v = 0; v < 8; ++v) {
                sf[v] = sigmoid_f(acc0[v]);
                so[v] = sigmoid_f(acc1[v]);
            }
        }

        WG_BARRIER_DS();   // B2: all A reads done, i*g products published

        if (w >= 8) {   // state update: c = sf*c + p ; h = so*tanh(c)
            const int blk = w - 8;
#pragma unroll
            for (int v = 0; v < 8; ++v) {
                const int m   = v + 8 * hi;
                const int col = blk * 16 + mr;
                const float p = exch[blk][m][mr];
                const float c = sf[v] * cstate[v] + p;
                cstate[v] = c;
                const float h = so[v] * tanh_f(c);
                hstate[v] = h;
                ssum[v] += h;
                ssq[v]  += h * h;
                Abuf[m * KTOT + I_ + col] = (_Float16)h;   // feedback for t+1
                out[((size_t)m * T_ + t) * H_ + col] = h;  // raw y
            }
        }
        // next-iteration staging writes only the x region; B1 orders h reads.
    }

    if (w >= 8) {
        const int blk = w - 8;
        // final h / c
#pragma unroll
        for (int v = 0; v < 8; ++v) {
            const int m   = v + 8 * hi;
            const int col = blk * 16 + mr;
            hn[m * H_ + col] = hstate[v];
            cn[m * H_ + col] = cstate[v];
        }
        // per-channel stats: reduce over the 8 m's, then lane<->lane+16
        float s = 0.f, q = 0.f;
#pragma unroll
        for (int v = 0; v < 8; ++v) { s += ssum[v]; q += ssq[v]; }
        s += __shfl_xor(s, 16, 32);
        q += __shfl_xor(q, 16, 32);
        if (hi == 0) {
            gsum[blk * 16 + mr]   = s;
            gsumsq[blk * 16 + mr] = q;
        }
    }
}

// ---------------------------------------------------------------------------
// Phase 2a: fold stats into per-channel scale/shift
// ---------------------------------------------------------------------------
__global__ void finalize_stats_kernel(const float* __restrict__ gsum,
                                      const float* __restrict__ gsumsq,
                                      const float* __restrict__ gamma,
                                      const float* __restrict__ beta,
                                      float* __restrict__ scale,
                                      float* __restrict__ shift)
{
    const int c = threadIdx.x;
    if (c < H_) {
        const float M    = (float)B_ * (float)T_;
        const float mean = gsum[c] / M;
        const float var  = gsumsq[c] / M - mean * mean;
        const float sc   = gamma[c] * rsqrtf(var + 1e-5f);
        scale[c] = sc;
        shift[c] = beta[c] - mean * sc;
    }
}

// ---------------------------------------------------------------------------
// Phase 2b: in-place BatchNorm + leaky-ReLU over y (128 MB r + 128 MB w,
// pure HBM-bandwidth bound: ~11 us at 23.3 TB/s)
// ---------------------------------------------------------------------------
__global__ void bn_lrelu_kernel(float* __restrict__ y,
                                const float* __restrict__ scale,
                                const float* __restrict__ shift)
{
    __shared__ float sS[H_], sB[H_];
    if (threadIdx.x < H_) {
        sS[threadIdx.x] = scale[threadIdx.x];
        sB[threadIdx.x] = shift[threadIdx.x];
    }
    __syncthreads();

    const size_t n4 = (size_t)B_ * T_ * H_ / 4;
    float4* y4 = (float4*)y;
    for (size_t i = (size_t)blockIdx.x * blockDim.x + threadIdx.x; i < n4;
         i += (size_t)gridDim.x * blockDim.x) {
        float4 v = y4[i];
        const int c = (int)((i * 4) & (H_ - 1));
        float r;
        r = v.x * sS[c + 0] + sB[c + 0]; v.x = r > 0.f ? r : 0.01f * r;
        r = v.y * sS[c + 1] + sB[c + 1]; v.y = r > 0.f ? r : 0.01f * r;
        r = v.z * sS[c + 2] + sB[c + 2]; v.z = r > 0.f ? r : 0.01f * r;
        r = v.w * sS[c + 3] + sB[c + 3]; v.w = r > 0.f ? r : 0.01f * r;
        y4[i] = v;
    }
}

// ---------------------------------------------------------------------------
extern "C" void kernel_launch(void* const* d_in, const int* in_sizes, int n_in,
                              void* d_out, int out_size, void* d_ws, size_t ws_size,
                              hipStream_t stream)
{
    const float* x     = (const float*)d_in[0];
    const float* Wih   = (const float*)d_in[1];
    const float* Whh   = (const float*)d_in[2];
    const float* bih   = (const float*)d_in[3];
    const float* bhh   = (const float*)d_in[4];
    const float* gamma = (const float*)d_in[5];
    const float* beta  = (const float*)d_in[6];

    float* y  = (float*)d_out;                    // [B,T,H]
    float* hn = y + (size_t)B_ * T_ * H_;         // [1,B,H]
    float* cn = hn + (size_t)B_ * H_;             // [1,B,H]

    float* gsum   = (float*)d_ws;                 // [H]
    float* gsumsq = gsum + H_;                    // [H]
    float* scale  = gsumsq + H_;                  // [H]
    float* shift  = scale + H_;                   // [H]

    lstm_scan_kernel<<<1, THREADS, 0, stream>>>(x, Wih, Whh, bih, bhh,
                                                y, hn, cn, gsum, gsumsq);
    finalize_stats_kernel<<<1, H_, 0, stream>>>(gsum, gsumsq, gamma, beta,
                                                scale, shift);
    bn_lrelu_kernel<<<4096, 256, 0, stream>>>(y, scale, shift);
}
